// MultiHeadSelfAttention_49976239456523
// MI455X (gfx1250) — matrix operand order repaired and checked by reading, not compiled, not run on hardware
//
#include <hip/hip_runtime.h>

// ---------------------------------------------------------------------------
// MultiHeadSelfAttention forward for MI455X (gfx1250, wave32, WMMA + TDM).
// All GEMMs + attention run through v_wmma_f32_16x16x32_bf16 (f32 accum).
// GEMM tiles are staged by the Tensor Data Mover (tensor_load_to_lds),
// double-buffered, with TENSORcnt-based pipelining.
// ---------------------------------------------------------------------------

typedef __attribute__((ext_vector_type(16))) __bf16 v16bf;
typedef __attribute__((ext_vector_type(8)))  __bf16 v8bf;
typedef __attribute__((ext_vector_type(8)))  float  v8f;
typedef __attribute__((ext_vector_type(4)))  float  f32x4;
typedef __attribute__((ext_vector_type(4)))  unsigned int u32x4;
typedef __attribute__((ext_vector_type(8)))  int i32x8;
typedef __attribute__((ext_vector_type(4)))  int i32x4;

#define EMBED 768
#define NHEAD 12
#define HDIM  64
#define BATCH 8
#define SEQ   1024
#define MTOT  (BATCH * SEQ)   // 8192

#if defined(__gfx1250__) && __has_builtin(__builtin_amdgcn_tensor_load_to_lds)
#define GEMM_USE_TDM 1
#else
#define GEMM_USE_TDM 0
#endif

__device__ __forceinline__ v8f vzero() {
  v8f z;
#pragma unroll
  for (int e = 0; e < 8; ++e) z[e] = 0.0f;
  return z;
}

__device__ __forceinline__ v16bf cat8(v8bf lo, v8bf hi) {
  return __builtin_shufflevector(lo, hi, 0, 1, 2, 3, 4, 5, 6, 7,
                                 8, 9, 10, 11, 12, 13, 14, 15);
}

__device__ __forceinline__ v8f wmma_bf16(v16bf a, v16bf b, v8f c) {
  // D = A(16x32 bf16) x B(32x16 bf16) + C(16x16 f32)
  return __builtin_amdgcn_wmma_f32_16x16x32_bf16(
      /*neg_a=*/false, a, /*neg_b=*/false, b,
      /*c_mod=*/(short)0, c, /*reuse_a=*/false, /*reuse_b=*/false);
}

#if GEMM_USE_TDM
// TDM 2D tile load: tile_w x tile_h (elements, 2B each) from a row-major
// tensor with row stride `stride_elems`, into LDS at `lds_addr`, with HW
// padding of 4 DWORDs (16B) after every 32 DWORDs (128B row) -> 144B rows.
// D# layout per CDNA5 ISA.
// This toolchain's builtin takes 6 args:
//   (u32x4 group0, i32x8 group1, i32x4 group2, i32x4 group3, i32x8, i32 cpol)
__device__ __forceinline__ void tdm_load_tile_2d(unsigned lds_addr,
                                                 const __bf16* gptr,
                                                 unsigned stride_elems,
                                                 unsigned tile_w,
                                                 unsigned tile_h) {
  unsigned long long ga = (unsigned long long)gptr;
  u32x4 g0;
  g0[0] = 1u;                                   // count=1, user descriptor
  g0[1] = lds_addr;                             // lds_addr (bytes)
  g0[2] = (unsigned)ga;                         // global_addr[31:0]
  g0[3] = (unsigned)((ga >> 32) & 0x01FFFFFFu)  // global_addr[56:32]
          | (2u << 30);                         // type=2 ("image")
  i32x8 g1;
  // data_size=2B (code 1), pad_enable=1, pad_interval code 4 (=32 DWORDs),
  // pad_amount code 3 (=4 DWORDs); workgroup_mask=0 (not in cluster).
  g1[0] = (int)((1u << 16) | (1u << 20) | (4u << 22) | (3u << 25));
  unsigned td0 = stride_elems;                  // tensor_dim0 (elements)
  unsigned td1 = 1u << 20;                      // tensor_dim1: large, no clip
  g1[1] = (int)((td0 & 0xFFFFu) << 16);         // [47:32]=atomic addr(0)
  g1[2] = (int)((td0 >> 16) | ((td1 & 0xFFFFu) << 16));
  g1[3] = (int)((td1 >> 16) | (tile_w << 16));  // tile_dim0
  g1[4] = (int)(tile_h & 0xFFFFu);              // tile_dim1 ; tile_dim2=0
  g1[5] = (int)stride_elems;                    // tensor_dim0_stride lo32
  g1[6] = 0;                                    // stride hi16 | dim1_stride
  g1[7] = 0;
  i32x4 z4;
  z4[0] = 0; z4[1] = 0; z4[2] = 0; z4[3] = 0;   // groups 2/3 unused (2D)
  i32x8 z8;
#pragma unroll
  for (int e = 0; e < 8; ++e) z8[e] = 0;
  __builtin_amdgcn_tensor_load_to_lds(g0, g1, z4, z4, z8, 0);
}

__device__ __forceinline__ unsigned lds_offset_of(const void* p) {
  // Generic pointers to LDS carry the LDS byte offset in their low 32 bits.
  return (unsigned)(unsigned long long)p;
}
#endif

// ---------------------------------------------------------------------------
// fp32 -> bf16 conversion (vectorized, 8 elements/thread)
// ---------------------------------------------------------------------------
__global__ void cvt_f32_to_bf16(const float* __restrict__ src,
                                __bf16* __restrict__ dst, int n8) {
  int i = blockIdx.x * blockDim.x + threadIdx.x;
  if (i >= n8) return;
  const f32x4* s = (const f32x4*)(src + (size_t)i * 8);
  f32x4 a = s[0];
  f32x4 b = s[1];
  v8bf o;
#pragma unroll
  for (int e = 0; e < 4; ++e) o[e] = (__bf16)a[e];
#pragma unroll
  for (int e = 0; e < 4; ++e) o[e + 4] = (__bf16)b[e];
  *(v8bf*)(dst + (size_t)i * 8) = o;
}

// ---------------------------------------------------------------------------
// Generic bf16 GEMM:  C[M,N] = scale * (A[M,K] @ W[N,K]^T) (+ bias)
// Block tile 128x128, BK=64, 8 waves (2x4), wave tile 64x32, 16 wmma/iter.
// Tiles staged by TDM into double-buffered LDS (row stride 72 halves = 144B,
// produced by TDM hardware padding). Fallback: manual staging.
// ---------------------------------------------------------------------------
template <bool OUT_F32, bool BIAS>
__global__ __launch_bounds__(256) void gemm_bf16_tn(
    const __bf16* __restrict__ A, const __bf16* __restrict__ W,
    void* __restrict__ Cvoid, const float* __restrict__ bias,
    int M, int Nout, int K, float scale) {
  __shared__ __align__(16) __bf16 a_tile[2][128 * 72];
  __shared__ __align__(16) __bf16 w_tile[2][128 * 72];

  const int t = threadIdx.x;
  const int lane = t & 31;
  const int wid = t >> 5;
  const int wm = wid & 1;   // 0..1 : 64-row slab
  const int wn = wid >> 1;  // 0..3 : 32-col slab
  const int m0 = blockIdx.y * 128;
  const int n0 = blockIdx.x * 128;
  const int l15 = lane & 15;
  const int upper = (lane >= 16);

  v8f acc[4][2];
#pragma unroll
  for (int i = 0; i < 4; ++i)
#pragma unroll
    for (int j = 0; j < 2; ++j) acc[i][j] = vzero();

  const int ahb = upper ? 8 : 0;    // A frag: K {0-7,16-23} / {8-15,24-31}
  const int bkb = upper ? 8 : 0;    // B frag: K {0-7,16-23} / {8-15,24-31}

  // 16 WMMA on one staged 128x64 x 64x128 tile pair.
  auto compute_tile = [&](int buf) {
#pragma unroll
    for (int kf = 0; kf < 2; ++kf) {
      v16bf af[4];
#pragma unroll
      for (int mf = 0; mf < 4; ++mf) {
        int r = wm * 64 + mf * 16 + l15;
        v8bf lo = *(const v8bf*)&a_tile[buf][r * 72 + kf * 32 + ahb];
        v8bf hi = *(const v8bf*)&a_tile[buf][r * 72 + kf * 32 + ahb + 16];
        af[mf] = cat8(lo, hi);
      }
      v16bf bfrag[2];
#pragma unroll
      for (int nf = 0; nf < 2; ++nf) {
        int c = wn * 32 + nf * 16 + l15;
        v8bf lo = *(const v8bf*)&w_tile[buf][c * 72 + kf * 32 + bkb];
        v8bf hi = *(const v8bf*)&w_tile[buf][c * 72 + kf * 32 + bkb + 16];
        bfrag[nf] = cat8(lo, hi);
      }
#pragma unroll
      for (int mf = 0; mf < 4; ++mf)
#pragma unroll
        for (int nf = 0; nf < 2; ++nf)
          acc[mf][nf] = wmma_bf16(af[mf], bfrag[nf], acc[mf][nf]);
    }
  };

#if GEMM_USE_TDM
  // ---- TDM double-buffered pipeline: wave 0 drives the DMA engine ----
  const bool issuer = (wid == 0);
  if (issuer) {
    tdm_load_tile_2d(lds_offset_of(&a_tile[0][0]),
                     A + (size_t)m0 * K, K, 64, 128);
    tdm_load_tile_2d(lds_offset_of(&w_tile[0][0]),
                     W + (size_t)n0 * K, K, 64, 128);
  }
  int buf = 0;
  for (int k0 = 0; k0 < K; k0 += 64, buf ^= 1) {
    if (issuer) {
      if (k0 + 64 < K) {
        tdm_load_tile_2d(lds_offset_of(&a_tile[buf ^ 1][0]),
                         A + (size_t)m0 * K + (k0 + 64), K, 64, 128);
        tdm_load_tile_2d(lds_offset_of(&w_tile[buf ^ 1][0]),
                         W + (size_t)n0 * K + (k0 + 64), K, 64, 128);
        // 2 DMAs still in flight (next buffer); current buffer complete.
        __builtin_amdgcn_s_wait_tensorcnt(2);
      } else {
        __builtin_amdgcn_s_wait_tensorcnt(0);
      }
    }
    __syncthreads();
    compute_tile(buf);
    __syncthreads();  // all reads done before DMA overwrites this buffer
  }
#else
  // ---- manual staging fallback (also taken by the host compile pass) ----
  const int srow = t >> 1;
  const int sseg = (t & 1) * 32;
  for (int k0 = 0; k0 < K; k0 += 64) {
    const __bf16* asrc = A + (size_t)(m0 + srow) * K + k0 + sseg;
    const __bf16* wsrc = W + (size_t)(n0 + srow) * K + k0 + sseg;
#pragma unroll
    for (int q = 0; q < 4; ++q) {
      *(v8bf*)&a_tile[0][srow * 72 + sseg + q * 8] = *(const v8bf*)(asrc + q * 8);
      *(v8bf*)&w_tile[0][srow * 72 + sseg + q * 8] = *(const v8bf*)(wsrc + q * 8);
    }
    if (k0 + 64 < K) {
      __builtin_prefetch(asrc + 64, 0, 0);
      __builtin_prefetch(wsrc + 64, 0, 0);
    }
    __syncthreads();
    compute_tile(0);
    __syncthreads();
  }
#endif

  // ---- epilogue (C layout: slot i -> row i / i+8, col = lane) ----
  const int rofs = upper ? 8 : 0;
#pragma unroll
  for (int mf = 0; mf < 4; ++mf)
#pragma unroll
    for (int nf = 0; nf < 2; ++nf) {
      int col = n0 + wn * 32 + nf * 16 + l15;
      float bv = 0.0f;
      if (BIAS) bv = bias[col];
#pragma unroll
      for (int i = 0; i < 8; ++i) {
        int row = m0 + wm * 64 + mf * 16 + i + rofs;
        float v = acc[mf][nf][i] * scale + bv;
        if (OUT_F32)
          ((float*)Cvoid)[(size_t)row * Nout + col] = v;
        else
          ((__bf16*)Cvoid)[(size_t)row * Nout + col] = (__bf16)v;
      }
    }
}

// ---------------------------------------------------------------------------
// Flash attention per (b,h). Each head's Q/K/V is a contiguous [1024,64]
// bf16 block (the reference's raw reshape makes this true). Q pre-scaled.
// Block = 256 threads (8 waves); each wave owns 16 query rows (Br=128),
// streams keys in Bc=64 chunks with online softmax.
// Output written to O in [B,N,H*D] (post-transpose) layout, bf16.
// ---------------------------------------------------------------------------
__global__ __launch_bounds__(256) void flash_attn_bf16(
    const __bf16* __restrict__ Q, const __bf16* __restrict__ Km,
    const __bf16* __restrict__ Vm, __bf16* __restrict__ O) {
  __shared__ __align__(16) __bf16 lds_k[64 * 72];        // [key][dim]
  __shared__ __align__(16) __bf16 lds_vt[64 * 72];       // [dim][key]
  __shared__ __align__(16) __bf16 lds_p[8 * 16 * 72];    // per-wave P tile

  const int bh = blockIdx.x >> 3;    // 0..95
  const int tile = blockIdx.x & 7;   // 0..7
  const int q0 = tile * 128;
  const int b = bh / NHEAD;
  const int h = bh % NHEAD;
  const int t = threadIdx.x;
  const int lane = t & 31;
  const int wid = t >> 5;
  const int l15 = lane & 15;
  const int upper = (lane >= 16);

  const __bf16* Qh = Q + (size_t)bh * (SEQ * HDIM);
  const __bf16* Kh = Km + (size_t)bh * (SEQ * HDIM);
  const __bf16* Vh = Vm + (size_t)bh * (SEQ * HDIM);

  // Q fragments for this wave's 16 rows (K = 64 -> 2 frags), kept in regs.
  v16bf qf[2];
  {
    int row = q0 + wid * 16 + l15;
    int hb = upper ? 8 : 0;
#pragma unroll
    for (int kf = 0; kf < 2; ++kf) {
      v8bf lo = *(const v8bf*)&Qh[(size_t)row * HDIM + kf * 32 + hb];
      v8bf hi = *(const v8bf*)&Qh[(size_t)row * HDIM + kf * 32 + hb + 16];
      qf[kf] = cat8(lo, hi);
    }
  }

  float mrow[8], lrow[8];
  v8f oacc[4];
#pragma unroll
  for (int i = 0; i < 8; ++i) { mrow[i] = -3.0e38f; lrow[i] = 0.0f; }
#pragma unroll
  for (int j = 0; j < 4; ++j) oacc[j] = vzero();

  const int bkb = upper ? 8 : 0;

  for (int c0 = 0; c0 < SEQ; c0 += 64) {
    // ---- stage K chunk [64x64] and V^T chunk ----
    {
      int key = t >> 2;
      int seg = (t & 3) * 16;
      const __bf16* ks = &Kh[(size_t)(c0 + key) * HDIM + seg];
      *(v8bf*)&lds_k[key * 72 + seg]     = *(const v8bf*)ks;
      *(v8bf*)&lds_k[key * 72 + seg + 8] = *(const v8bf*)(ks + 8);
      const __bf16* vs = &Vh[(size_t)(c0 + key) * HDIM + seg];
#pragma unroll
      for (int j = 0; j < 16; ++j) lds_vt[(seg + j) * 72 + key] = vs[j];
      if (c0 + 64 < SEQ) {
        __builtin_prefetch(ks + 64 * HDIM, 0, 0);
        __builtin_prefetch(vs + 64 * HDIM, 0, 0);
      }
    }
    __syncthreads();

    // ---- S = Q @ K^T  (Q pre-scaled by 1/sqrt(D)) ----
    v8f sacc[4];
#pragma unroll
    for (int j = 0; j < 4; ++j) sacc[j] = vzero();
#pragma unroll
    for (int nf = 0; nf < 4; ++nf) {
      int key = nf * 16 + l15;
#pragma unroll
      for (int kf = 0; kf < 2; ++kf) {
        v8bf lo = *(const v8bf*)&lds_k[key * 72 + kf * 32 + bkb];
        v8bf hi = *(const v8bf*)&lds_k[key * 72 + kf * 32 + bkb + 16];
        sacc[nf] = wmma_bf16(qf[kf], cat8(lo, hi), sacc[nf]);
      }
    }

    // ---- online softmax (rows i / i+8 per half-wave; butterfly over 16) --
#pragma unroll
    for (int i = 0; i < 8; ++i) {
      float mx = fmaxf(fmaxf(sacc[0][i], sacc[1][i]),
                       fmaxf(sacc[2][i], sacc[3][i]));
      for (int off = 8; off >= 1; off >>= 1)
        mx = fmaxf(mx, __shfl_xor(mx, off, 32));
      float mnew = fmaxf(mrow[i], mx);
      float alpha = __expf(mrow[i] - mnew);
      mrow[i] = mnew;
      float rs = 0.0f;
#pragma unroll
      for (int j = 0; j < 4; ++j) {
        float p = __expf(sacc[j][i] - mnew);
        sacc[j][i] = p;
        rs += p;
      }
      for (int off = 8; off >= 1; off >>= 1) rs += __shfl_xor(rs, off, 32);
      lrow[i] = lrow[i] * alpha + rs;
#pragma unroll
      for (int j = 0; j < 4; ++j) oacc[j][i] *= alpha;
    }

    // ---- P (C-layout regs) -> wave-private LDS -> A-fragments ----
    __bf16* myp = lds_p + wid * (16 * 72);
    {
      int rbase = upper ? 8 : 0;
#pragma unroll
      for (int j = 0; j < 4; ++j) {
        int col = j * 16 + l15;
#pragma unroll
        for (int i = 0; i < 8; ++i)
          myp[(rbase + i) * 72 + col] = (__bf16)sacc[j][i];
      }
    }
    asm volatile("" ::: "memory");  // keep ds loads after ds stores
    v16bf pf[2];
    {
      int hb = upper ? 8 : 0;
#pragma unroll
      for (int kf = 0; kf < 2; ++kf) {
        v8bf lo = *(const v8bf*)&myp[l15 * 72 + kf * 32 + hb];
        v8bf hi = *(const v8bf*)&myp[l15 * 72 + kf * 32 + hb + 16];
        pf[kf] = cat8(lo, hi);
      }
    }

    // ---- O += P @ V  (B from transposed V: contiguous K per lane) ----
#pragma unroll
    for (int nf = 0; nf < 4; ++nf) {
      int dim = nf * 16 + l15;
#pragma unroll
      for (int kf = 0; kf < 2; ++kf) {
        v8bf lo = *(const v8bf*)&lds_vt[dim * 72 + kf * 32 + bkb];
        v8bf hi = *(const v8bf*)&lds_vt[dim * 72 + kf * 32 + bkb + 16];
        oacc[nf] = wmma_bf16(pf[kf], cat8(lo, hi), oacc[nf]);
      }
    }
    __syncthreads();
  }

  // ---- normalize and write to [B,N,H*D] layout ----
#pragma unroll
  for (int nf = 0; nf < 4; ++nf) {
    int dim = nf * 16 + l15;
#pragma unroll
    for (int i = 0; i < 8; ++i) {
      int row = q0 + wid * 16 + i + (upper ? 8 : 0);
      float v = oacc[nf][i] / lrow[i];
      O[(size_t)b * (SEQ * EMBED) + (size_t)row * EMBED + h * HDIM + dim] =
          (__bf16)v;
    }
  }
}

// ---------------------------------------------------------------------------
// Launch
// ---------------------------------------------------------------------------
extern "C" void kernel_launch(void* const* d_in, const int* in_sizes, int n_in,
                              void* d_out, int out_size, void* d_ws,
                              size_t ws_size, hipStream_t stream) {
  (void)in_sizes; (void)n_in; (void)out_size; (void)ws_size;

  const float* x   = (const float*)d_in[0];
  const float* w_q = (const float*)d_in[1];
  const float* w_k = (const float*)d_in[2];
  const float* w_v = (const float*)d_in[3];
  const float* w_o = (const float*)d_in[4];
  const float* b_o = (const float*)d_in[5];

  // Workspace carve-up (~68 MB total, 256B aligned chunks).
  char* ws = (char*)d_ws;
  size_t off = 0;
  auto carve = [&](size_t bytes) -> void* {
    void* p = ws + off;
    off += (bytes + 255) & ~(size_t)255;
    return p;
  };
  const size_t xeN = (size_t)MTOT * EMBED;   // 6,291,456
  const size_t weN = (size_t)EMBED * EMBED;  //   589,824
  __bf16* xb  = (__bf16*)carve(xeN * 2);
  __bf16* qb  = (__bf16*)carve(xeN * 2);
  __bf16* kb  = (__bf16*)carve(xeN * 2);
  __bf16* vb  = (__bf16*)carve(xeN * 2);
  __bf16* ob  = (__bf16*)carve(xeN * 2);
  __bf16* wqb = (__bf16*)carve(weN * 2);
  __bf16* wkb = (__bf16*)carve(weN * 2);
  __bf16* wvb = (__bf16*)carve(weN * 2);
  __bf16* wob = (__bf16*)carve(weN * 2);

  // 1) fp32 -> bf16 conversions
  {
    int n8x = (int)(xeN / 8);
    cvt_f32_to_bf16<<<(n8x + 255) / 256, 256, 0, stream>>>(x, xb, n8x);
    int n8w = (int)(weN / 8);
    dim3 g((n8w + 255) / 256);
    cvt_f32_to_bf16<<<g, 256, 0, stream>>>(w_q, wqb, n8w);
    cvt_f32_to_bf16<<<g, 256, 0, stream>>>(w_k, wkb, n8w);
    cvt_f32_to_bf16<<<g, 256, 0, stream>>>(w_v, wvb, n8w);
    cvt_f32_to_bf16<<<g, 256, 0, stream>>>(w_o, wob, n8w);
  }

  // 2) QKV projections (Q pre-scaled by 1/sqrt(HDIM) = 0.125)
  dim3 gg(EMBED / 128, MTOT / 128);  // (6, 64)
  gemm_bf16_tn<false, false><<<gg, 256, 0, stream>>>(
      xb, wqb, qb, nullptr, MTOT, EMBED, EMBED, 0.125f);
  gemm_bf16_tn<false, false><<<gg, 256, 0, stream>>>(
      xb, wkb, kb, nullptr, MTOT, EMBED, EMBED, 1.0f);
  gemm_bf16_tn<false, false><<<gg, 256, 0, stream>>>(
      xb, wvb, vb, nullptr, MTOT, EMBED, EMBED, 1.0f);

  // 3) attention (96 heads x 8 query tiles of 128 rows)
  flash_attn_bf16<<<96 * 8, 256, 0, stream>>>(qb, kb, vb, ob);

  // 4) output projection with bias, fp32 result
  gemm_bf16_tn<true, true><<<gg, 256, 0, stream>>>(
      ob, wob, d_out, b_o, MTOT, EMBED, EMBED, 1.0f);
}
